// LatentAlignmentDecoder_71734543778043
// MI455X (gfx1250) — matrix operand order repaired and checked by reading, not compiled, not run on hardware
//
#include <hip/hip_runtime.h>
#include <hip/hip_bf16.h>

// ---------------------------------------------------------------------------
// LatentAlignmentDecoder on MI455X (gfx1250): wave32, WMMA bf16,
// TDM tensor_load_to_lds + async global->LDS staging, persistent-WG LSTM.
// B=16, S=128, H=256, NH=8, HD=32, F=512, V=30522
// ---------------------------------------------------------------------------

#define B_  16
#define S_  128
#define H_  256
#define F_  512
#define V_  30522
#define G4_ 1024           // 4*H
#define ROWS_ (B_ * S_)    // 2048
#define HP_ 264            // padded LDS row (bf16 elems) for the LSTM h tile
#define BROW_ 144          // padded LDS row (bytes) for staged B tiles

typedef __bf16 bf16;
typedef __attribute__((ext_vector_type(16))) __bf16 v16bf;
typedef __attribute__((ext_vector_type(8)))  __bf16 v8bf;
typedef __attribute__((ext_vector_type(8)))  float  v8f;
typedef __attribute__((ext_vector_type(4)))  unsigned int v4u;
typedef __attribute__((ext_vector_type(8)))  int v8i;
typedef __attribute__((ext_vector_type(4)))  int v4i;

#define DEV static __device__ __forceinline__

DEV v8bf load8(const bf16* p) { return *(const v8bf*)p; }

DEV v16bf combine16(v8bf lo, v8bf hi) {
  v16bf r;
#pragma unroll
  for (int i = 0; i < 8; ++i) { r[i] = lo[i]; r[i + 8] = hi[i]; }
  return r;
}

DEV v8f wmma_bf16f32(v16bf a, v16bf b, v8f c) {
  // D = A(16x32 bf16) * B(32x16 bf16) + C(16x16 f32)
  return __builtin_amdgcn_wmma_f32_16x16x32_bf16(
      false, a, false, b, (short)0, c, false, false);
}

DEV float sigmoidf_(float x) { return 1.0f / (1.0f + __expf(-x)); }

// async copy of one 16B chunk: global -> LDS (ASYNCcnt-tracked)
DEV void async_copy_b128(const void* gsrc, void* lds_dst) {
  unsigned lds = (unsigned)(size_t)lds_dst;   // low 32 bits = LDS offset
  asm volatile("global_load_async_to_lds_b128 %0, %1, off"
               :: "v"(lds), "v"(gsrc)
               : "memory");
}

DEV void wait_async_all() {
  asm volatile("s_wait_asynccnt 0" ::: "memory");
}

// ---------------------------------------------------------------------------
// TDM: one descriptor-driven DMA of a 64x64 bf16 tile (row stride K elems)
// into LDS with 16B padding appended per 128B row (-> BROW_=144B rows).
// D# layout per CDNA5 ISA 8.3-8.5. Issued by one wave; TENSORcnt-tracked.
// This toolchain exposes the 6-arg builtin: (v4u, v8i, v4i, v4i, v8i, cpol).
// ---------------------------------------------------------------------------
DEV void tdm_load_tile(const bf16* gsrc, void* lds_dst, int K, int rows_left) {
  unsigned long long ga = (unsigned long long)(size_t)gsrc;
  v4u g0;
  g0[0] = 1u;                                      // count=1 (valid), user mode
  g0[1] = (unsigned)(size_t)lds_dst;               // lds_addr (bytes)
  g0[2] = (unsigned)ga;                            // global_addr[31:0]
  g0[3] = (unsigned)((ga >> 32) & 0x01FFFFFFu)     // global_addr[56:32]
        | (2u << 30);                              // type=2 ("image")
  v8i g1;
  g1[0] = (int)((1u << 16)                         // data_size = 2 bytes
              | (1u << 20)                         // pad_enable
              | (4u << 22)                         // pad_interval: 32 DWORDs
              | (3u << 25));                       // pad_amount: 4 DWORDs
  g1[1] = (int)(((unsigned)K & 0xFFFFu) << 16);    // tensor_dim0[15:0]
  g1[2] = (int)(((unsigned)rows_left & 0xFFFFu) << 16); // tensor_dim1[15:0]
  g1[3] = (int)(64u << 16);                        // tile_dim0 = 64
  g1[4] = 64;                                      // tile_dim1 = 64, tile_dim2=0
  g1[5] = K;                                       // tensor_dim0_stride[31:0]
  g1[6] = 0;                                       // stride hi / dim1_stride lo
  g1[7] = 0;
  v4i g2; g2[0] = 0; g2[1] = 0; g2[2] = 0; g2[3] = 0;
  v4i g3; g3[0] = 0; g3[1] = 0; g3[2] = 0; g3[3] = 0;
  v8i g4;
#pragma unroll
  for (int i = 0; i < 8; ++i) g4[i] = 0;
  __builtin_amdgcn_tensor_load_to_lds(g0, g1, g2, g3, g4, 0);
}

// ---------------------------------------------------------------------------
// elementwise f32 -> bf16 convert
// ---------------------------------------------------------------------------
__global__ void k_f32_to_bf16(const float* __restrict__ in,
                              bf16* __restrict__ out, int n) {
  int i = blockIdx.x * blockDim.x + threadIdx.x;
  if (i < n) out[i] = (bf16)in[i];
}

// ---------------------------------------------------------------------------
// embedding gather: out[row,:] = bf16(emb[ids[row],:])
// ---------------------------------------------------------------------------
__global__ void k_embed_gather(const int* __restrict__ ids,
                               const float* __restrict__ emb,
                               bf16* __restrict__ out) {
  int row = blockIdx.x;               // 0..2047
  int id  = ids[row];
  const float* src = emb + (size_t)id * H_;
  bf16* dst = out + (size_t)row * H_;
  for (int i = threadIdx.x; i < H_; i += blockDim.x) dst[i] = (bf16)src[i];
}

// ---------------------------------------------------------------------------
// xg[b*S+t, n] -> xgT[(t*G4+n)*16 + b]  (WMMA C-layout friendly)
// ---------------------------------------------------------------------------
__global__ void k_repack_xg(const float* __restrict__ in,
                            float* __restrict__ out) {
  int i = blockIdx.x * blockDim.x + threadIdx.x;   // < 2048*1024
  int b = i & 15;
  int n = (i >> 4) & (G4_ - 1);
  int t = i >> 14;
  out[i] = in[(size_t)((b << 7) + t) * G4_ + n];
}

// ---------------------------------------------------------------------------
// WMMA GEMM with staged-B LDS double buffering:
//   C[M,N] = A[M,K](bf16 row-major) * W[N,K]^T + b0 + b1
// Block = 256 thr (8 waves). Block tile: 128(M) x 64(N); K stages of 64.
// use_tdm=1: wave 0 issues one TENSOR_LOAD_TO_LDS per stage (TDM pads rows).
// use_tdm=0: all lanes issue global_load_async_to_lds_b128 chunks (ASYNCcnt).
// Requires M%128==0, K%64==0. N tail: clamped loads / predicated stores.
// ---------------------------------------------------------------------------
__global__ void k_gemm_bf16(const bf16* __restrict__ A,
                            const bf16* __restrict__ W,
                            const float* __restrict__ b0,
                            const float* __restrict__ b1,
                            float* __restrict__ C,
                            bf16* __restrict__ Cbf,
                            int M, int N, int K, int use_tdm) {
  __shared__ __align__(16) unsigned char shB[2][64 * BROW_];

  const int tid  = threadIdx.x;
  const int lane = tid & 31;
  const int lhi  = lane >> 4;
  const int lm   = lane & 15;
  const int w    = tid >> 5;

  const int nMblk = M >> 7;
  const int mblk  = blockIdx.x % nMblk;
  const int nt4   = blockIdx.x / nMblk;
  const int m0    = (mblk << 7) + (w << 4);
  const int n0    = nt4 << 6;

  // stage one 64x64 bf16 B tile (K-chunk kb) into dst
  auto stage = [&](int kb, unsigned char* dst) {
    if (use_tdm) {
      if (w == 0) tdm_load_tile(W + (size_t)n0 * K + kb, dst, K, N - n0);
    } else {
#pragma unroll
      for (int c = tid; c < 512; c += 256) {
        int row = c >> 3;
        int off = (c & 7) << 4;
        int n = n0 + row; if (n > N - 1) n = N - 1;
        const unsigned char* g =
            (const unsigned char*)(W + (size_t)n * K + kb) + off;
        async_copy_b128(g, dst + row * BROW_ + off);
      }
    }
  };

  v8f acc[4];
#pragma unroll
  for (int t = 0; t < 4; ++t)
#pragma unroll
    for (int r = 0; r < 8; ++r) acc[t][r] = 0.0f;

  const bf16* Arow = A + (size_t)(m0 + lm) * K;

  stage(0, shB[0]);
  // prime A fragment registers for kb=0
  v8bf a0 = load8(Arow + 8 * lhi);
  v8bf a1 = load8(Arow + 16 + 8 * lhi);

  int buf = 0;
  for (int kb = 0; kb < K; kb += 64) {
    wait_async_all();                        // no-op on the TDM path
    __builtin_amdgcn_s_wait_tensorcnt(0);    // no-op on the async path
    __syncthreads();                         // staged tile visible to all
    if (kb + 64 < K) stage(kb + 64, shB[buf ^ 1]);

#pragma unroll
    for (int kk = 0; kk < 64; kk += 32) {
      v16bf af = combine16(a0, a1);
      int kn = kb + kk + 32;
      if (kn < K) {                   // register double-buffer next A chunk
        a0 = load8(Arow + kn + 8 * lhi);
        a1 = load8(Arow + kn + 16 + 8 * lhi);
      }
#pragma unroll
      for (int nt = 0; nt < 4; ++nt) {
        const bf16* Brow = (const bf16*)(shB[buf] +
            ((nt << 4) + lm) * BROW_ + ((kk + 8 * lhi) << 1));
        v16bf bfrag = combine16(load8(Brow), load8(Brow + 16));
        acc[nt] = wmma_bf16f32(af, bfrag, acc[nt]);
      }
    }
    buf ^= 1;
  }

  // epilogue: C/D layout -> row = m0 + r + 8*lhi, col = n0 + nt*16 + lm
#pragma unroll
  for (int nt = 0; nt < 4; ++nt) {
    int col = n0 + (nt << 4) + lm;
    if (col < N) {
      float bias = (b0 ? b0[col] : 0.0f) + (b1 ? b1[col] : 0.0f);
#pragma unroll
      for (int r = 0; r < 8; ++r) {
        size_t idx = (size_t)(m0 + r + 8 * lhi) * N + col;
        float v = acc[nt][r] + bias;
        C[idx] = v;
        if (Cbf) Cbf[idx] = (bf16)v;
      }
    }
  }
}

// ---------------------------------------------------------------------------
// LSTM layer recurrence (persistent single workgroup, 16 waves = 512 thr).
// xgT[t][n][b] already = x@Wih.T + bih + bhh, in WMMA C layout.
// Wave w owns hidden slice j in [16w,16w+16): 4 WMMA N-tiles (i,f,g,o).
// Cell state c lives in accumulator-layout registers; h shared via LDS bf16
// (rows padded to 264 elems -> conflict-free A-fragment reads).
// ---------------------------------------------------------------------------
__global__ void k_lstm_layer(const float* __restrict__ xgT,
                             const bf16* __restrict__ Whh,
                             bf16* __restrict__ y) {
  __shared__ bf16 hsh[16 * HP_];

  const int tid  = threadIdx.x;
  const int lane = tid & 31;
  const int lhi  = lane >> 4;
  const int lm   = lane & 15;
  const int w    = tid >> 5;          // wave 0..15
  const int j0   = w << 4;            // hidden slice base

  for (int i = tid; i < 16 * HP_; i += blockDim.x) hsh[i] = (bf16)0.0f;
  __syncthreads();

  v8f c;
#pragma unroll
  for (int r = 0; r < 8; ++r) c[r] = 0.0f;

  for (int t = 0; t < S_; ++t) {
    v8f acc[4];
#pragma unroll
    for (int g = 0; g < 4; ++g) {
      int n = g * H_ + j0 + lm;
      acc[g] = *(const v8f*)(xgT + ((size_t)t * G4_ + n) * 16 + 8 * lhi);
    }

    for (int kb = 0; kb < H_; kb += 32) {
      const bf16* hrow = hsh + lm * HP_ + kb;
      v16bf af = combine16(load8(hrow + 8 * lhi),
                           load8(hrow + 16 + 8 * lhi));
#pragma unroll
      for (int g = 0; g < 4; ++g) {
        int n = g * H_ + j0 + lm;
        const bf16* Wrow = Whh + (size_t)n * H_ + kb + 8 * lhi;
        v16bf bfrag = combine16(load8(Wrow), load8(Wrow + 16));
        acc[g] = wmma_bf16f32(af, bfrag, acc[g]);
      }
    }

    __syncthreads();                  // everyone done reading h(t-1)

#pragma unroll
    for (int r = 0; r < 8; ++r) {
      float ig = sigmoidf_(acc[0][r]);
      float fg = sigmoidf_(acc[1][r]);
      float gg = tanhf(acc[2][r]);
      float og = sigmoidf_(acc[3][r]);
      float cc = fg * c[r] + ig * gg;
      c[r] = cc;
      float hh = og * tanhf(cc);
      int b = r + 8 * lhi;
      int j = j0 + lm;
      bf16 hb = (bf16)hh;
      hsh[b * HP_ + j] = hb;
      y[(size_t)(b * S_ + t) * H_ + j] = hb;
    }
    __syncthreads();                  // h(t) visible before next step reads
  }
}

// ---------------------------------------------------------------------------
// Multi-head cross attention: q[2048,256] vs k,v[256,256] (16 clusters/batch)
// grid = B*NH blocks, block = S threads (one q position per thread).
// ---------------------------------------------------------------------------
__global__ void k_attention(const float* __restrict__ q,
                            const float* __restrict__ k,
                            const float* __restrict__ v,
                            bf16* __restrict__ ctxbf) {
  const int b = blockIdx.x >> 3;
  const int h = blockIdx.x & 7;
  const int s = threadIdx.x;

  const float* qp = q + (size_t)(b * S_ + s) * H_ + h * 32;
  float qr[32];
#pragma unroll
  for (int d = 0; d < 32; ++d) qr[d] = qp[d];

  float sc[16], mx = -1e30f;
  for (int c = 0; c < 16; ++c) {
    const float* kp = k + (size_t)(b * 16 + c) * H_ + h * 32;
    float dot = 0.0f;
#pragma unroll
    for (int d = 0; d < 32; ++d) dot += qr[d] * kp[d];
    sc[c] = dot * 0.17677669529663689f;   // 1/sqrt(32)
    mx = fmaxf(mx, sc[c]);
  }
  float sum = 0.0f;
  for (int c = 0; c < 16; ++c) { sc[c] = __expf(sc[c] - mx); sum += sc[c]; }
  float inv = 1.0f / sum;

  float ctx[32];
#pragma unroll
  for (int d = 0; d < 32; ++d) ctx[d] = 0.0f;
  for (int c = 0; c < 16; ++c) {
    const float* vp = v + (size_t)(b * 16 + c) * H_ + h * 32;
    float a = sc[c] * inv;
#pragma unroll
    for (int d = 0; d < 32; ++d) ctx[d] += a * vp[d];
  }
  bf16* op = ctxbf + (size_t)(b * S_ + s) * H_ + h * 32;
#pragma unroll
  for (int d = 0; d < 32; ++d) op[d] = (bf16)ctx[d];
}

// ---------------------------------------------------------------------------
// per-row logsumexp over V logits (grid = 2048 rows)
// ---------------------------------------------------------------------------
__global__ void k_row_logsumexp(const float* __restrict__ logits,
                                float* __restrict__ logZ) {
  const int row = blockIdx.x;
  const float* p = logits + (size_t)row * V_;
  __shared__ float sm[256];

  float m = -1e30f;
  for (int i = threadIdx.x; i < V_; i += 256) m = fmaxf(m, p[i]);
  sm[threadIdx.x] = m;
  __syncthreads();
  for (int s = 128; s > 0; s >>= 1) {
    if (threadIdx.x < s) sm[threadIdx.x] = fmaxf(sm[threadIdx.x], sm[threadIdx.x + s]);
    __syncthreads();
  }
  m = sm[0];
  __syncthreads();

  float sum = 0.0f;
  for (int i = threadIdx.x; i < V_; i += 256) sum += __expf(p[i] - m);
  sm[threadIdx.x] = sum;
  __syncthreads();
  for (int s = 128; s > 0; s >>= 1) {
    if (threadIdx.x < s) sm[threadIdx.x] += sm[threadIdx.x + s];
    __syncthreads();
  }
  if (threadIdx.x == 0) logZ[row] = m + __logf(sm[0]);
}

// ---------------------------------------------------------------------------
// CE loss on shifted tokens (single block)
// ---------------------------------------------------------------------------
__global__ void k_loss(const float* __restrict__ logits,
                       const float* __restrict__ logZ,
                       const int* __restrict__ ids,
                       float* __restrict__ out_loss) {
  __shared__ float sm[256];
  float acc = 0.0f;
  const int count = B_ * (S_ - 1);    // 2032
  for (int idx = threadIdx.x; idx < count; idx += 256) {
    int b = idx / (S_ - 1);
    int t = idx % (S_ - 1);
    int row = b * S_ + t;
    int tgt = ids[b * S_ + t + 1];
    acc += logits[(size_t)row * V_ + tgt] - logZ[row];
  }
  sm[threadIdx.x] = acc;
  __syncthreads();
  for (int s = 128; s > 0; s >>= 1) {
    if (threadIdx.x < s) sm[threadIdx.x] += sm[threadIdx.x + s];
    __syncthreads();
  }
  if (threadIdx.x == 0) out_loss[0] = -sm[0] / (float)count;
}

// ---------------------------------------------------------------------------
// host-side launcher
// ---------------------------------------------------------------------------
static inline int cdiv(int a, int b) { return (a + b - 1) / b; }

extern "C" void kernel_launch(void* const* d_in, const int* in_sizes, int n_in,
                              void* d_out, int out_size, void* d_ws, size_t ws_size,
                              hipStream_t stream) {
  (void)in_sizes; (void)n_in; (void)out_size; (void)ws_size;

  const int*   ids  = (const int*)  d_in[0];
  const float* cf   = (const float*)d_in[1];
  const float* emb  = (const float*)d_in[2];
  const float* Wc   = (const float*)d_in[3];
  const float* bc   = (const float*)d_in[4];
  const float* Wih0 = (const float*)d_in[5];
  const float* Whh0 = (const float*)d_in[6];
  const float* bih0 = (const float*)d_in[7];
  const float* bhh0 = (const float*)d_in[8];
  const float* Wih1 = (const float*)d_in[9];
  const float* Whh1 = (const float*)d_in[10];
  const float* bih1 = (const float*)d_in[11];
  const float* bhh1 = (const float*)d_in[12];
  const float* Wq   = (const float*)d_in[13];
  const float* Wk   = (const float*)d_in[14];
  const float* Wv   = (const float*)d_in[15];
  const float* Wo   = (const float*)d_in[16];
  const float* bq   = (const float*)d_in[17];
  const float* bk   = (const float*)d_in[18];
  const float* bv   = (const float*)d_in[19];
  const float* bo   = (const float*)d_in[20];
  const float* Wout = (const float*)d_in[21];
  const float* bout = (const float*)d_in[22];

  float* out    = (float*)d_out;
  float* logits = out + 1;            // (loss, logits) concatenated

  // ---- workspace bump allocator -------------------------------------------
  char* wsp = (char*)d_ws;
  auto alloc = [&](size_t bytes) -> void* {
    void* p = (void*)wsp;
    wsp += (bytes + 255) & ~(size_t)255;
    return p;
  };
  bf16*  web    = (bf16*) alloc((size_t)ROWS_ * H_ * 2);      // word embeds
  bf16*  cfb    = (bf16*) alloc((size_t)B_ * 16 * F_ * 2);
  bf16*  wcb    = (bf16*) alloc((size_t)H_ * F_ * 2);
  float* cef    = (float*)alloc((size_t)B_ * 16 * H_ * 4);
  bf16*  ceb    = (bf16*) alloc((size_t)B_ * 16 * H_ * 2);
  bf16*  wih0b  = (bf16*) alloc((size_t)G4_ * H_ * 2);
  bf16*  whh0b  = (bf16*) alloc((size_t)G4_ * H_ * 2);
  bf16*  wih1b  = (bf16*) alloc((size_t)G4_ * H_ * 2);
  bf16*  whh1b  = (bf16*) alloc((size_t)G4_ * H_ * 2);
  float* xg     = (float*)alloc((size_t)ROWS_ * G4_ * 4);     // reused L0/L1
  float* xgT    = (float*)alloc((size_t)ROWS_ * G4_ * 4);     // repacked
  bf16*  x1b    = (bf16*) alloc((size_t)ROWS_ * H_ * 2);
  bf16*  lob    = (bf16*) alloc((size_t)ROWS_ * H_ * 2);
  bf16*  wqb    = (bf16*) alloc((size_t)H_ * H_ * 2);
  bf16*  wkb    = (bf16*) alloc((size_t)H_ * H_ * 2);
  bf16*  wvb    = (bf16*) alloc((size_t)H_ * H_ * 2);
  bf16*  wob    = (bf16*) alloc((size_t)H_ * H_ * 2);
  float* qf     = (float*)alloc((size_t)ROWS_ * H_ * 4);
  float* kf     = (float*)alloc((size_t)B_ * 16 * H_ * 4);
  float* vf     = (float*)alloc((size_t)B_ * 16 * H_ * 4);
  bf16*  ctxb   = (bf16*) alloc((size_t)ROWS_ * H_ * 2);
  float* alf    = (float*)alloc((size_t)ROWS_ * H_ * 4);
  bf16*  alb    = (bf16*) alloc((size_t)ROWS_ * H_ * 2);
  bf16*  woutb  = (bf16*) alloc((size_t)V_ * H_ * 2);
  float* logZ   = (float*)alloc((size_t)ROWS_ * 4);

  // ---- weight / input conversions to bf16 ---------------------------------
  auto cvt = [&](const float* src, bf16* dst, int n) {
    k_f32_to_bf16<<<cdiv(n, 256), 256, 0, stream>>>(src, dst, n);
  };
  cvt(cf,   cfb,   B_ * 16 * F_);
  cvt(Wc,   wcb,   H_ * F_);
  cvt(Wih0, wih0b, G4_ * H_);
  cvt(Whh0, whh0b, G4_ * H_);
  cvt(Wih1, wih1b, G4_ * H_);
  cvt(Whh1, whh1b, G4_ * H_);
  cvt(Wq,   wqb,   H_ * H_);
  cvt(Wk,   wkb,   H_ * H_);
  cvt(Wv,   wvb,   H_ * H_);
  cvt(Wo,   wob,   H_ * H_);
  cvt(Wout, woutb, V_ * H_);

  k_embed_gather<<<ROWS_, 256, 0, stream>>>(ids, emb, web);

  auto gemm = [&](const bf16* A, const bf16* W, const float* b0, const float* b1,
                  float* C, bf16* Cbf, int M, int N, int K) {
    int blocks = (M >> 7) * ((N + 63) >> 6);
    int use_tdm = (N >= 512) ? 1 : 0;     // TDM for wide GEMMs, async for narrow
    k_gemm_bf16<<<blocks, 256, 0, stream>>>(A, W, b0, b1, C, Cbf, M, N, K, use_tdm);
  };

  // cluster embeds: [256,512] x [256,512]^T
  gemm(cfb, wcb, bc, nullptr, cef, ceb, B_ * 16, H_, F_);

  // layer 0: xg = we @ Wih0^T + bih0 + bhh0 ; repack ; recurrence
  gemm(web, wih0b, bih0, bhh0, xg, nullptr, ROWS_, G4_, H_);
  k_repack_xg<<<(ROWS_ * G4_) / 256, 256, 0, stream>>>(xg, xgT);
  k_lstm_layer<<<1, 512, 0, stream>>>(xgT, whh0b, x1b);

  // layer 1
  gemm(x1b, wih1b, bih1, bhh1, xg, nullptr, ROWS_, G4_, H_);
  k_repack_xg<<<(ROWS_ * G4_) / 256, 256, 0, stream>>>(xg, xgT);
  k_lstm_layer<<<1, 512, 0, stream>>>(xgT, whh1b, lob);

  // q / k / v projections
  gemm(lob, wqb, bq, nullptr, qf, nullptr, ROWS_, H_, H_);
  gemm(ceb, wkb, bk, nullptr, kf, nullptr, B_ * 16, H_, H_);
  gemm(ceb, wvb, bv, nullptr, vf, nullptr, B_ * 16, H_, H_);

  // attention
  k_attention<<<B_ * 8, S_, 0, stream>>>(qf, kf, vf, ctxb);

  // output alignment + logits
  gemm(ctxb, wob, bo, nullptr, alf, alb, ROWS_, H_, H_);
  gemm(alb, woutb, bout, nullptr, logits, nullptr, ROWS_, V_, H_);

  // loss
  k_row_logsumexp<<<ROWS_, 256, 0, stream>>>(logits, logZ);
  k_loss<<<1, 256, 0, stream>>>(logits, logZ, ids, out);
}
